// Decoder_10239202034039
// MI455X (gfx1250) — matrix operand order repaired and checked by reading, not compiled, not run on hardware
//
#include <hip/hip_runtime.h>
#include <hip/hip_bf16.h>
#include <math.h>

// ---------------------------------------------------------------------------
// MI455X (gfx1250) decoder implementation.
// All large einsums run through v_wmma_f32_16x16x32_f16 (wave32 WMMA), with
// f32 -> f16 conversion fused into the LDS staging and f32 accumulation.
// Tile: 128(M) x 256(N) x 64(K) per 256-thread block; 8 waves in a 2x4 grid,
// each wave owns a 64x64 sub-tile = 16 f32 accumulators -> 32 WMMAs per
// staging round per wave (2 K-substeps of 32).
// ---------------------------------------------------------------------------

typedef _Float16 v16h __attribute__((ext_vector_type(16)));
typedef _Float16 v8h  __attribute__((ext_vector_type(8)));
typedef _Float16 v4h  __attribute__((ext_vector_type(4)));
typedef float    v8f  __attribute__((ext_vector_type(8)));

static constexpr int Hn    = 8;
static constexpr int Dn    = 1024;
static constexpr int Sn    = 1024;
static constexpr int MIDn  = 2048;
static constexpr int SOUTn = 512;

#define LDSP 72   // LDS row stride in halves for BK=64 (144B: 16B aligned, bank-spread)

// ---------------------------------------------------------------------------
// Tiled WMMA GEMM:  C[M][N] = act( op_relu(A) * B )
//   TA = 0: A stored row-major [M][K] (lda = K-stride)
//   TA = 1: A stored [K][M]            (lda = M-stride)  (used for k^T q, cat^T)
//   TB = 0: B stored [N][K]            (ldb = K-stride)  (weights stored (N,K))
//   TB = 1: B stored [K][N]            (ldb = N-stride)
//   RA = 1: relu applied to A while converting to f16
//   ACT = 1: tanh on output
// Requires M % 128 == 0, N % 256 == 0, K % 64 == 0 (true for all call sites).
// ---------------------------------------------------------------------------
template<int TA, int TB, int RA, int ACT>
__global__ __launch_bounds__(256)
void gemm_wmma_kernel(const float* __restrict__ A, const float* __restrict__ B,
                      float* __restrict__ C, int M, int N, int K,
                      int lda, int ldb, int ldc)
{
    (void)M; (void)N;
    __shared__ _Float16 Ah[128 * LDSP];   // 18 KB
    __shared__ _Float16 Bh[256 * LDSP];   // 36 KB

    const int tid  = threadIdx.x;
    const int lane = tid & 31;
    const int wid  = tid >> 5;
    const int wm   = (wid >> 2) * 64;   // wave M offset in tile (0,64)
    const int wn   = (wid & 3)  * 64;   // wave N offset in tile (0..192)
    const int lm   = lane & 15;
    const int lh   = lane >> 4;
    const int m0   = blockIdx.y * 128;
    const int n0   = blockIdx.x * 256;

    v8f acc[4][4];
#pragma unroll
    for (int i = 0; i < 4; ++i)
#pragma unroll
        for (int j = 0; j < 4; ++j)
#pragma unroll
            for (int e = 0; e < 8; ++e) acc[i][j][e] = 0.0f;

    for (int k0 = 0; k0 < K; k0 += 64) {
        // ---- stage A tile (128x64) -> LDS [m][k] as f16 (optional relu) ----
        if (TA == 0) {
#pragma unroll
            for (int p = 0; p < 8; ++p) {
                int f   = tid + p * 256;       // 2048 float4 per tile
                int row = f >> 4;
                int c4  = (f & 15) << 2;
                float4 v = *(const float4*)(A + (size_t)(m0 + row) * lda + k0 + c4);
                if (RA) { v.x = fmaxf(v.x, 0.f); v.y = fmaxf(v.y, 0.f);
                          v.z = fmaxf(v.z, 0.f); v.w = fmaxf(v.w, 0.f); }
                v4h h; h[0] = (_Float16)v.x; h[1] = (_Float16)v.y;
                       h[2] = (_Float16)v.z; h[3] = (_Float16)v.w;
                *(v4h*)&Ah[row * LDSP + c4] = h;
            }
        } else {
#pragma unroll
            for (int p = 0; p < 32; ++p) {
                int idx = tid + p * 256;
                int m   = idx & 127;           // contiguous along m -> coalesced
                int k   = idx >> 7;            // 0..63
                float x = A[(size_t)(k0 + k) * lda + m0 + m];
                if (RA) x = fmaxf(x, 0.f);
                Ah[m * LDSP + k] = (_Float16)x;
            }
        }
        // ---- stage B tile (256x64) -> LDS [n][k] as f16 ----
        if (TB == 0) {
#pragma unroll
            for (int p = 0; p < 16; ++p) {
                int f   = tid + p * 256;       // 4096 float4 per tile
                int row = f >> 4;
                int c4  = (f & 15) << 2;
                float4 v = *(const float4*)(B + (size_t)(n0 + row) * ldb + k0 + c4);
                v4h h; h[0] = (_Float16)v.x; h[1] = (_Float16)v.y;
                       h[2] = (_Float16)v.z; h[3] = (_Float16)v.w;
                *(v4h*)&Bh[row * LDSP + c4] = h;
            }
        } else {
#pragma unroll
            for (int p = 0; p < 64; ++p) {
                int idx = tid + p * 256;
                int n   = idx & 255;           // contiguous along n -> coalesced
                int k   = idx >> 8;            // 0..63
                Bh[n * LDSP + k] = (_Float16)B[(size_t)(k0 + k) * ldb + n0 + n];
            }
        }
        __syncthreads();

        // ---- two K-substeps of 32: gather fragments per ISA §7.12.2 ----
#pragma unroll
        for (int ks = 0; ks < 64; ks += 32) {
            v16h af[4], bf[4];
#pragma unroll
            for (int i = 0; i < 4; ++i) {
                // A 16x32: lane holds row lm; K chunks [ks+lh*8, +8) and +16
                const _Float16* p = &Ah[(wm + i * 16 + lm) * LDSP + ks + lh * 8];
                v8h lo = *(const v8h*)p;
                v8h hi = *(const v8h*)(p + 16);
                v16h a;
#pragma unroll
                for (int e = 0; e < 8; ++e) { a[e] = lo[e]; a[e + 8] = hi[e]; }
                af[i] = a;
            }
#pragma unroll
            for (int j = 0; j < 4; ++j) {
                // B 32x16: lane holds col lm; K chunks [ks+lh*8, +8) and +16
                const _Float16* p = &Bh[(wn + j * 16 + lm) * LDSP + ks + lh * 8];
                v8h lo = *(const v8h*)p;
                v8h hi = *(const v8h*)(p + 16);
                v16h b;
#pragma unroll
                for (int e = 0; e < 8; ++e) { b[e] = lo[e]; b[e + 8] = hi[e]; }
                bf[j] = b;
            }
#pragma unroll
            for (int i = 0; i < 4; ++i)
#pragma unroll
                for (int j = 0; j < 4; ++j)
                    acc[i][j] = __builtin_amdgcn_wmma_f32_16x16x32_f16(
                        false, af[i], false, bf[j], (short)0, acc[i][j], false, false);
        }
        __syncthreads();
    }

    // ---- write back: C/D layout VGPR r -> row r + lh*8, col lm ----
#pragma unroll
    for (int i = 0; i < 4; ++i)
#pragma unroll
        for (int j = 0; j < 4; ++j)
#pragma unroll
            for (int r = 0; r < 8; ++r) {
                int row = m0 + wm + i * 16 + lh * 8 + r;
                int col = n0 + wn + j * 16 + lm;
                float v = acc[i][j][r];
                if (ACT == 1) v = tanhf(v);
                C[(size_t)row * ldc + col] = v;
            }
}

// ---------------------------------------------------------------------------
// Fused residual + LayerNorm over S, reading the (S,D)-major MHA output
// transposed:  out[d][s] = LN_s( inp[d][s] + mt[s][d] ) * gamma[s] + beta[s]
// ---------------------------------------------------------------------------
__global__ __launch_bounds__(256)
void add_ln_t_kernel(float* __restrict__ out, const float* __restrict__ inp,
                     const float* __restrict__ mt, const float* __restrict__ gamma,
                     const float* __restrict__ beta, int Dd, int Ss)
{
    __shared__ float rs[256];
    __shared__ float rq[256];
    const int d = blockIdx.x;
    const int tid = threadIdx.x;
    float s = 0.f, q = 0.f;
    for (int i = tid; i < Ss; i += 256) {
        float v = inp[(size_t)d * Ss + i] + mt[(size_t)i * Dd + d];
        s += v; q += v * v;
    }
    rs[tid] = s; rq[tid] = q;
    __syncthreads();
    for (int off = 128; off > 0; off >>= 1) {
        if (tid < off) { rs[tid] += rs[tid + off]; rq[tid] += rq[tid + off]; }
        __syncthreads();
    }
    const float mu  = rs[0] / (float)Ss;
    const float var = rq[0] / (float)Ss - mu * mu;
    const float inv = rsqrtf(var + 1e-6f);
    for (int i = tid; i < Ss; i += 256) {
        float v = inp[(size_t)d * Ss + i] + mt[(size_t)i * Dd + d];
        out[(size_t)d * Ss + i] = (v - mu) * inv * gamma[i] + beta[i];
    }
}

// ---------------------------------------------------------------------------
// Host-side dispatch
// ---------------------------------------------------------------------------
static inline void launch_gemm(int ta, int tb, int ra, int act,
                               const float* A, const float* B, float* C,
                               int M, int N, int K, int lda, int ldb, int ldc,
                               hipStream_t s)
{
    dim3 grid(N / 256, M / 128), blk(256);
    if      (ta == 0 && tb == 1 && ra == 0 && act == 0)
        gemm_wmma_kernel<0,1,0,0><<<grid, blk, 0, s>>>(A, B, C, M, N, K, lda, ldb, ldc);
    else if (ta == 0 && tb == 0 && ra == 1 && act == 0)
        gemm_wmma_kernel<0,0,1,0><<<grid, blk, 0, s>>>(A, B, C, M, N, K, lda, ldb, ldc);
    else if (ta == 1 && tb == 1 && ra == 0 && act == 0)
        gemm_wmma_kernel<1,1,0,0><<<grid, blk, 0, s>>>(A, B, C, M, N, K, lda, ldb, ldc);
    else if (ta == 1 && tb == 0 && ra == 1 && act == 0)
        gemm_wmma_kernel<1,0,1,0><<<grid, blk, 0, s>>>(A, B, C, M, N, K, lda, ldb, ldc);
    else if (ta == 0 && tb == 0 && ra == 1 && act == 1)
        gemm_wmma_kernel<0,0,1,1><<<grid, blk, 0, s>>>(A, B, C, M, N, K, lda, ldb, ldc);
}

// One MHA block: per head (q,k,v head-MLPs, k^T q, v*b), then concat proj.
// Writes m1t as (S, D) row-major (transposed residual), consumed by LN kernel.
static void run_mha(const float* q, const float* k, const float* v,
                    const float* W1, const float* W2,
                    const float* cw1, const float* cw2,
                    float* zbuf, float* sbuf, float* bmat,
                    float* cat, float* ybuf, float* m1t, hipStream_t s)
{
    const float* qkv[3] = { q, k, v };
    const size_t DS = (size_t)Dn * Sn;
    for (int h = 0; h < Hn; ++h) {
        for (int qi = 0; qi < 3; ++qi) {
            const float* W1h = W1 + (size_t)(h * 3 + qi) * MIDn * Sn;   // (MID,S)  NK
            const float* W2h = W2 + (size_t)(h * 3 + qi) * Sn * MIDn;   // (S,MID)  NK
            // z = relu(qkv) * W1^T  : (D,MID)
            launch_gemm(0,0,1,0, qkv[qi], W1h, zbuf, Dn, MIDn, Sn, Sn, Sn, MIDn, s);
            // s = relu(z) * W2^T    : (D,S)
            launch_gemm(0,0,1,0, zbuf, W2h, sbuf + (size_t)qi * DS, Dn, Sn, MIDn, MIDn, MIDn, Sn, s);
        }
        const float* sq = sbuf;            // s[h][0]
        const float* sk = sbuf + DS;       // s[h][1]
        const float* sv = sbuf + 2 * DS;   // s[h][2]
        // b[i][j] = sum_d k[d][i] q[d][j]  : A=(KM) k, B=(KN) q
        launch_gemm(1,1,0,0, sk, sq, bmat, Sn, Sn, Dn, Sn, Sn, Sn, s);
        // att[d][t] = sum_s v[d][s] b[s][t] : A=(MK) v, B=(KN) b
        launch_gemm(0,1,0,0, sv, bmat, cat + (size_t)h * DS, Dn, Sn, Sn, Sn, Sn, Sn, s);
    }
    // y[s][m] = sum_c relu(cat[c][s]) cw1[m][c]  : A=(KM) cat, B=(NK) cw1
    launch_gemm(1,0,1,0, cat, cw1, ybuf, Sn, MIDn, Hn * Dn, Sn, Hn * Dn, MIDn, s);
    // m1t[s][d] = sum_m relu(y[s][m]) cw2[d][m]  : (S,D) row-major
    launch_gemm(0,0,1,0, ybuf, cw2, m1t, Sn, Dn, MIDn, MIDn, MIDn, Dn, s);
}

extern "C" void kernel_launch(void* const* d_in, const int* in_sizes, int n_in,
                              void* d_out, int out_size, void* d_ws, size_t ws_size,
                              hipStream_t stream)
{
    (void)in_sizes; (void)n_in; (void)out_size; (void)ws_size;

    const float* inp    = (const float*)d_in[0];
    const float* enc_k  = (const float*)d_in[1];
    const float* enc_v  = (const float*)d_in[2];
    const float* w_qkv1 = (const float*)d_in[3];
    const float* w_qkv2 = (const float*)d_in[4];
    const float* mh1_W1 = (const float*)d_in[5];
    const float* mh1_W2 = (const float*)d_in[6];
    const float* mh2_W1 = (const float*)d_in[7];
    const float* mh2_W2 = (const float*)d_in[8];
    const float* c1_w1  = (const float*)d_in[9];
    const float* c1_w2  = (const float*)d_in[10];
    const float* c2_w1  = (const float*)d_in[11];
    const float* c2_w2  = (const float*)d_in[12];
    const float* l1_w1  = (const float*)d_in[13];
    const float* l1_w2  = (const float*)d_in[14];
    const float* l2_w1  = (const float*)d_in[15];
    const float* l2_w2  = (const float*)d_in[16];
    const float* gamma  = (const float*)d_in[17];
    const float* beta   = (const float*)d_in[18];
    float* out = (float*)d_out;

    const size_t DS = (size_t)Dn * Sn;
    float* ws = (float*)d_ws;
    size_t off = 0;
    auto alloc = [&](size_t n) { float* p = ws + off; off += n; return p; };
    // Workspace (~103 MB of f32):
    float* qkv   = alloc(3 * DS);               // q/k/v activations
    float* zbuf  = alloc((size_t)Dn * MIDn);    // head-MLP hidden
    float* sbuf  = alloc(3 * DS);               // per-head q,k,v outputs
    float* bmat  = alloc((size_t)Sn * Sn);      // k^T q
    float* cat   = alloc((size_t)Hn * DS);      // concat of per-head attention
    float* ybuf  = alloc((size_t)Sn * MIDn);    // concat projection hidden
    float* m1t   = alloc((size_t)Sn * Dn);      // MHA output, (S,D)-major
    float* h1    = alloc(DS);
    float* h2    = alloc(DS);
    float* xbuf  = alloc((size_t)Dn * MIDn);    // head MLP (reused)
    float* x2buf = alloc((size_t)Dn * SOUTn);

    // ---- block 1 ----
    for (int qi = 0; qi < 3; ++qi)   // qkv1[q] = w_qkv1[q] (MK) * inp (KN)
        launch_gemm(0,1,0,0, w_qkv1 + (size_t)qi * Dn * Dn, inp,
                    qkv + (size_t)qi * DS, Dn, Sn, Dn, Dn, Sn, Sn, stream);
    run_mha(qkv, qkv + DS, qkv + 2 * DS, mh1_W1, mh1_W2, c1_w1, c1_w2,
            zbuf, sbuf, bmat, cat, ybuf, m1t, stream);
    add_ln_t_kernel<<<Dn, 256, 0, stream>>>(h1, inp, m1t, gamma, beta, Dn, Sn);

    // ---- block 2 (cross-attention: k,v from encoder) ----
    launch_gemm(0,1,0,0, w_qkv2, h1, qkv, Dn, Sn, Dn, Dn, Sn, Sn, stream);
    run_mha(qkv, enc_k, enc_v, mh2_W1, mh2_W2, c2_w1, c2_w2,
            zbuf, sbuf, bmat, cat, ybuf, m1t, stream);
    add_ln_t_kernel<<<Dn, 256, 0, stream>>>(h2, inp, m1t, gamma, beta, Dn, Sn);

    // ---- output head ----
    launch_gemm(0,0,1,0, h2,    l1_w1, xbuf,  Dn, MIDn,  Sn,    Sn,    Sn,    MIDn,  stream);
    launch_gemm(0,0,1,0, xbuf,  l1_w2, x2buf, Dn, SOUTn, MIDn,  MIDn,  MIDn,  SOUTn, stream);
    launch_gemm(0,0,1,0, x2buf, l2_w1, xbuf,  Dn, MIDn,  SOUTn, SOUTn, SOUTn, MIDn,  stream);
    launch_gemm(0,0,1,1, xbuf,  l2_w2, out,   Dn, SOUTn, MIDn,  MIDn,  MIDn,  SOUTn, stream);
}
